// MultiHeadAttentionLayer_50766513439004
// MI455X (gfx1250) — matrix operand order repaired and checked by reading, not compiled, not run on hardware
//
#include <hip/hip_runtime.h>

typedef __attribute__((ext_vector_type(16))) _Float16 v16h;
typedef __attribute__((ext_vector_type(8)))  _Float16 v8h;
typedef __attribute__((ext_vector_type(8)))  float    v8f;

#define IN_DIM    128
#define QKV_DIM   384   // Q(128) | K(128) | V(128) packed
#define NUM_HEADS 8
#define HEAD_DIM  16
#define NCT       24    // column tiles of 16 over QKV_DIM
#define WPK_ELEMS (NCT * 4 * 32 * 16)   // 49152 f16 = 96 KB

// ---------------- h -> f16 ----------------
__global__ void gat_cvt_h(const float* __restrict__ h, _Float16* __restrict__ hf, int n) {
  int i = blockIdx.x * blockDim.x + threadIdx.x;
  if (i < n) hf[i] = (_Float16)h[i];
}

// ---------------- pack Wq|Wk|Wv into WMMA-B fragment-linear f16 layout ----------------
// Layout: Wpk[((ct*4 + kk)*32 + lane)*16 + e]
//   column  n = ct*16 + (lane & 15)
//   k       = kk*32 + (e<8 ? (lane>>4)*8 + e : 16 + (lane>>4)*8 + (e-8))
//             (lanes 0-15 K {0..7,16..23}, lanes 16-31 K {8..15,24..31}, matches A)
__global__ void gat_pack_w(const float* __restrict__ Wq, const float* __restrict__ Wk,
                           const float* __restrict__ Wv, _Float16* __restrict__ Wpk) {
  int idx = blockIdx.x * blockDim.x + threadIdx.x;
  if (idx >= WPK_ELEMS) return;
  int e    = idx & 15;
  int lane = (idx >> 4) & 31;
  int kk   = (idx >> 9) & 3;
  int ct   = idx >> 11;
  int k    = kk * 32 + ((e < 8) ? ((lane >> 4) * 8 + e) : (16 + (lane >> 4) * 8 + (e - 8)));
  int col  = ct * 16 + (lane & 15);
  float w  = (col < 128) ? Wq[k * 128 + col]
           : (col < 256) ? Wk[k * 128 + (col - 128)]
                         : Wv[k * 128 + (col - 256)];
  Wpk[idx] = (_Float16)w;
}

__global__ void gat_pack_b(const float* __restrict__ bq, const float* __restrict__ bk,
                           const float* __restrict__ bv, float* __restrict__ bpk) {
  int c = blockIdx.x * blockDim.x + threadIdx.x;
  if (c >= QKV_DIM) return;
  bpk[c] = (c < 128) ? bq[c] : (c < 256) ? bk[c - 128] : bv[c - 256];
}

#if __has_builtin(__builtin_amdgcn_tensor_load_to_lds) && __has_builtin(__builtin_amdgcn_s_wait_tensorcnt)
#define USE_TDM 1
typedef unsigned int u32x4 __attribute__((ext_vector_type(4)));
typedef int          i32x4 __attribute__((ext_vector_type(4)));
typedef int          i32x8 __attribute__((ext_vector_type(8)));
#endif

// ---------------- QKV projection: [N,128] x [128,384] via v_wmma_f32_16x16x32_f16 ----
// Block = 8 waves = 8 row tiles (128 rows). The 96KB B panel is staged in LDS once per
// block by the Tensor Data Mover, then every wave streams all 24 column tiles from LDS
// while its 4 A-fragments stay resident in VGPRs.
__global__ __launch_bounds__(256) void gat_qkv_wmma(const _Float16* __restrict__ hf,
                                                    const _Float16* __restrict__ Wpk,
                                                    const float* __restrict__ bpk,
                                                    float* __restrict__ QKV, int nRows) {
  __shared__ _Float16 Bsh[WPK_ELEMS];   // 96 KB of the WGP's 320 KB LDS

  const int wave = threadIdx.x >> 5;
  const int lane = threadIdx.x & 31;
  const int rt   = blockIdx.x * 8 + wave;   // row tile per wave
  const int kh   = lane >> 4;               // K-half select (A layout, ISA 7.12.2)
  const int m    = lane & 15;               // A row within tile / D column within tile

#if defined(USE_TDM)
  if (threadIdx.x < 32) {                   // one wave issues the TDM stage
    const unsigned lds_addr = (unsigned)(size_t)(&Bsh[0]);
    const unsigned long long ga = (unsigned long long)(size_t)Wpk;
    // D# group 0: count=1 | lds_addr | global_addr[56:0] | type=2
    u32x4 g0;
    g0[0] = 1u;
    g0[1] = lds_addr;
    g0[2] = (unsigned)(ga & 0xffffffffull);
    g0[3] = (unsigned)((ga >> 32) & 0x01ffffffull) | (2u << 30);
    // D# group 1: wg_mask=0, data_size=1 (2B), 1-row tile of 49152 elements
    i32x8 g1;
    g1[0] = (int)(1u << 16);                              // data_size = 2 bytes
    g1[1] = (int)((WPK_ELEMS & 0xffff) << 16);            // tensor_dim0[15:0]
    g1[2] = (int)(((WPK_ELEMS >> 16) & 0xffff) | (1 << 16)); // dim0 hi | tensor_dim1=1
    g1[3] = (int)((WPK_ELEMS & 0xffff) << 16);            // tile_dim0 = 49152
    g1[4] = 1;                                            // tile_dim1=1, tile_dim2=0
    g1[5] = WPK_ELEMS;                                    // tensor_dim0_stride lo32
    g1[6] = 0;
    g1[7] = 0;
    i32x4 gz = {0, 0, 0, 0};
#if defined(__clang_major__) && __clang_major__ >= 23
    i32x8 gz8 = {0, 0, 0, 0, 0, 0, 0, 0};
    __builtin_amdgcn_tensor_load_to_lds(g0, g1, gz, gz, gz8, 0);
#else
    __builtin_amdgcn_tensor_load_to_lds(g0, g1, gz, gz, 0);
#endif
    __builtin_amdgcn_s_wait_tensorcnt(0);                 // s_wait_tensorcnt 0
  }
#else
  for (int i = threadIdx.x; i < WPK_ELEMS / 8; i += 256)
    *(v8h*)(&Bsh[i * 8]) = *(const v8h*)(Wpk + (size_t)i * 8);
#endif
  __syncthreads();

  int arow = rt * 16 + m;
  if (arow >= nRows) arow = nRows - 1;      // clamp (stores are guarded)

  // Keep all 4 K-step A fragments resident (ISA 7.12.2 16-bit A layout):
  // lanes 0-15 hold K {kk*32+0..7, +16..23}; lanes 16-31 hold {+8..15, +24..31}
  v16h afr[4];
#pragma unroll
  for (int kk = 0; kk < 4; ++kk) {
    const _Float16* ap = hf + (size_t)arow * IN_DIM + kk * 32 + kh * 8;
    v8h lo = *(const v8h*)(ap);             // global_load_b128
    v8h hi = *(const v8h*)(ap + 16);
#pragma unroll
    for (int i = 0; i < 8; ++i) { afr[kk][i] = lo[i]; afr[kk][8 + i] = hi[i]; }
  }

  for (int ct = 0; ct < NCT; ++ct) {
    v8f acc = {};
#pragma unroll
    for (int kk = 0; kk < 4; ++kk) {
      v16h b = *(const v16h*)(&Bsh[((ct * 4 + kk) * 32 + lane) << 4]); // ds_load_b128 x2
      acc = __builtin_amdgcn_wmma_f32_16x16x32_f16(
          /*neg_a=*/false, afr[kk], /*neg_b=*/false, b,
          /*c_mod=*/(short)0, acc, /*reuse_a=*/false, /*reuse_b=*/false);
    }
    // D layout: VGPR r -> M = r + 8*(lane>>4), N = lane&15
    const int col  = ct * 16 + m;
    const float bb = bpk[col];
#pragma unroll
    for (int r = 0; r < 8; ++r) {
      const int orow = rt * 16 + r + 8 * kh;
      if (orow < nRows) QKV[(size_t)orow * QKV_DIM + col] = acc[r] + bb;
    }
  }
}

// ---------------- edge phase: wave per edge, 4-lane head groups ----------------
__global__ __launch_bounds__(256) void gat_edge(const int* __restrict__ src,
                                                const int* __restrict__ dst,
                                                const float* __restrict__ QKV,
                                                float* __restrict__ out,   // wV accum [N,128]
                                                float* __restrict__ z,     // [N,8]
                                                int nEdges) {
  const int e = blockIdx.x * 8 + (threadIdx.x >> 5);
  if (e >= nEdges) return;
  const int lane = threadIdx.x & 31;
  const int s = src[e];
  const int d = dst[e];
  // contiguous 512B bursts per wave (all L2-resident: QKV=76.8MB, out=25.6MB << 192MB L2)
  const float4 kv = *(const float4*)(QKV + (size_t)s * QKV_DIM + 128 + lane * 4);
  const float4 qv = *(const float4*)(QKV + (size_t)d * QKV_DIM +       lane * 4);
  float p = kv.x * qv.x + kv.y * qv.y + kv.z * qv.z + kv.w * qv.w;
  p += __shfl_xor(p, 1);
  p += __shfl_xor(p, 2);               // all 4 lanes of a head group hold the head score
  const float sc = __expf(fminf(fmaxf(p * 0.25f, -5.0f), 5.0f));  // scale = 1/sqrt(16)
  const float4 vv = *(const float4*)(QKV + (size_t)s * QKV_DIM + 256 + lane * 4);
  float* op = out + (size_t)d * 128 + lane * 4;
  atomicAdd(op + 0, vv.x * sc);
  atomicAdd(op + 1, vv.y * sc);
  atomicAdd(op + 2, vv.z * sc);
  atomicAdd(op + 3, vv.w * sc);
  if ((lane & 3) == 0) atomicAdd(z + (size_t)d * 8 + (lane >> 2), sc);
}

// ---------------- normalize ----------------
__global__ void gat_norm(float* __restrict__ out, const float* __restrict__ z, int total) {
  int i = blockIdx.x * blockDim.x + threadIdx.x;
  if (i >= total) return;
  const int n  = i >> 7;
  const int hh = (i >> 4) & 7;
  out[i] = out[i] / z[n * 8 + hh];
}

extern "C" void kernel_launch(void* const* d_in, const int* in_sizes, int n_in,
                              void* d_out, int out_size, void* d_ws, size_t ws_size,
                              hipStream_t stream) {
  const float* h   = (const float*)d_in[0];
  const int*   src = (const int*)  d_in[1];
  const int*   dst = (const int*)  d_in[2];
  const float* Wq  = (const float*)d_in[3];
  const float* Wk  = (const float*)d_in[4];
  const float* Wv  = (const float*)d_in[5];
  const float* bq  = (const float*)d_in[6];
  const float* bk  = (const float*)d_in[7];
  const float* bv  = (const float*)d_in[8];

  const int N = in_sizes[0] / IN_DIM;   // 50000
  const int E = in_sizes[1];            // 800000

  // workspace layout (16B aligned)
  char* ws = (char*)d_ws;
  size_t off = 0;
  _Float16* hf  = (_Float16*)(ws + off); off += (size_t)N * IN_DIM * sizeof(_Float16);
  _Float16* Wpk = (_Float16*)(ws + off); off += (size_t)WPK_ELEMS * sizeof(_Float16);
  float*    bpk = (float*)   (ws + off); off += (size_t)QKV_DIM * sizeof(float) + 16;
  off = (off + 15) & ~(size_t)15;
  float*    QKV = (float*)   (ws + off); off += (size_t)N * QKV_DIM * sizeof(float);
  float*    z   = (float*)   (ws + off); off += (size_t)N * NUM_HEADS * sizeof(float);

  float* out = (float*)d_out;

  hipMemsetAsync(out, 0, (size_t)out_size * sizeof(float), stream);
  hipMemsetAsync(z,   0, (size_t)N * NUM_HEADS * sizeof(float), stream);

  {
    const int total = N * IN_DIM;
    gat_cvt_h<<<(total + 255) / 256, 256, 0, stream>>>(h, hf, total);
  }
  gat_pack_w<<<(WPK_ELEMS + 255) / 256, 256, 0, stream>>>(Wq, Wk, Wv, Wpk);
  gat_pack_b<<<2, 256, 0, stream>>>(bq, bk, bv, bpk);

  {
    const int nRowTiles  = (N + 15) / 16;          // 3125
    const int nBlocks    = (nRowTiles + 7) / 8;    // 391
    gat_qkv_wmma<<<nBlocks, 256, 0, stream>>>(hf, Wpk, bpk, QKV, N);
  }

  gat_edge<<<(E + 7) / 8, 256, 0, stream>>>(src, dst, QKV, out, z, E);

  {
    const int total = N * NUM_HEADS * HEAD_DIM;
    gat_norm<<<(total + 255) / 256, 256, 0, stream>>>(out, z, total);
  }
}
